// GAT_4672924418049
// MI455X (gfx1250) — compile-verified, matrix operand order checked by reading, not run on hardware
//
#include <hip/hip_runtime.h>
#include <hip/hip_bf16.h>

typedef __attribute__((ext_vector_type(16))) __bf16 v16bf;
typedef __attribute__((ext_vector_type(8)))  float  v8f;

static constexpr int   NN = 32768;     // nodes
static constexpr int   NE = 262144;    // edges
static constexpr int   NG = 64;        // graphs
static constexpr float NEG_SLOPE = 0.2f;

// ---------------- helpers ----------------

__device__ __forceinline__ unsigned short f2bf(float x) {
  unsigned int b = __float_as_uint(x);
  b += 0x7FFFu + ((b >> 16) & 1u);          // round-to-nearest-even
  return (unsigned short)(b >> 16);
}

__device__ __forceinline__ void atomicMaxF32(float* addr, float val) {
  // sign-aware monotonic mapping of IEEE754 ordering onto int/uint atomics
  if (__float_as_int(val) >= 0)
    atomicMax((int*)addr, __float_as_int(val));
  else
    atomicMin((unsigned int*)addr, __float_as_uint(val));
}

__global__ void fill_f32(float* __restrict__ p, float v, int n) {
  int i = blockIdx.x * blockDim.x + threadIdx.x;
  if (i < n) p[i] = v;
}

__global__ void conv_bf16(const float* __restrict__ x, unsigned short* __restrict__ y, int n) {
  int i = blockIdx.x * blockDim.x + threadIdx.x;
  if (i < n) y[i] = f2bf(x[i]);
}

// W [K, NC] (row-major) -> Wt [NC, K] bf16 (K contiguous)
__global__ void convT_bf16(const float* __restrict__ W, unsigned short* __restrict__ Wt,
                           int K, int NC) {
  int i = blockIdx.x * blockDim.x + threadIdx.x;
  if (i >= K * NC) return;
  int k = i / NC, n = i % NC;
  Wt[(size_t)n * K + k] = f2bf(W[i]);
}

// ---------------- bf16 WMMA GEMM: C[NN,512] = A[NN,K] x B[K,512] ----------------
// A bf16 row-major [NN,K]; Bt bf16 [512,K] (pre-transposed). Block = 8 waves,
// block tile 128 rows x 64 cols; each wave: 16 rows x 64 cols via 4 accumulators.
__global__ __launch_bounds__(256)
void gat_gemm_bf16(const unsigned short* __restrict__ A,
                   const unsigned short* __restrict__ Bt,
                   float* __restrict__ C, int K) {
  const int lane = threadIdx.x & 31;
  const int wave = threadIdx.x >> 5;
  const int hi   = lane >> 4;      // 0: K{0..7,16..23}, 1: K{8..15,24..31}
  const int l16  = lane & 15;
  const int row0 = blockIdx.x * 128 + wave * 16;
  const int col0 = blockIdx.y * 64;

  const unsigned short* aRow = A  + (size_t)(row0 + l16) * K + hi * 8;
  const unsigned short* bRow = Bt + (size_t)(col0 + l16) * K + hi * 8;

  v8f acc0 = {}, acc1 = {}, acc2 = {}, acc3 = {};
  for (int k0 = 0; k0 < K; k0 += 32) {
    union Frag { v16bf v; uint4 u[2]; } af, b0, b1, b2, b3;
    const unsigned short* ap = aRow + k0;
    af.u[0] = *(const uint4*)(ap);
    af.u[1] = *(const uint4*)(ap + 16);
    const unsigned short* bp = bRow + k0;
    b0.u[0] = *(const uint4*)(bp);               b0.u[1] = *(const uint4*)(bp + 16);
    b1.u[0] = *(const uint4*)(bp + 16 * K);      b1.u[1] = *(const uint4*)(bp + 16 * K + 16);
    b2.u[0] = *(const uint4*)(bp + 32 * K);      b2.u[1] = *(const uint4*)(bp + 32 * K + 16);
    b3.u[0] = *(const uint4*)(bp + 48 * K);      b3.u[1] = *(const uint4*)(bp + 48 * K + 16);
    acc0 = __builtin_amdgcn_wmma_f32_16x16x32_bf16(false, af.v, false, b0.v, (short)0, acc0, false, false);
    acc1 = __builtin_amdgcn_wmma_f32_16x16x32_bf16(false, af.v, false, b1.v, (short)0, acc1, false, false);
    acc2 = __builtin_amdgcn_wmma_f32_16x16x32_bf16(false, af.v, false, b2.v, (short)0, acc2, false, false);
    acc3 = __builtin_amdgcn_wmma_f32_16x16x32_bf16(false, af.v, false, b3.v, (short)0, acc3, false, false);
  }
  // C/D layout: VGPR r, lanes 0-15 -> row r, lanes 16-31 -> row r+8; col = lane&15
#pragma unroll
  for (int r = 0; r < 8; ++r) {
    float* cp = C + (size_t)(row0 + r + 8 * hi) * 512 + col0 + l16;
    cp[0]  = acc0[r];
    cp[16] = acc1[r];
    cp[32] = acc2[r];
    cp[48] = acc3[r];
  }
}

// ---------------- GAT layers 0-2 (H=8, D=64) ----------------

__global__ void gat_scores(const float* __restrict__ ft, const float* __restrict__ al,
                           const float* __restrict__ ar, float* __restrict__ el,
                           float* __restrict__ er) {
  int t = blockIdx.x * blockDim.x + threadIdx.x;   // NN*8
  if (t >= NN * 8) return;
  int n = t >> 3, h = t & 7;
  const float* f = ft + (size_t)n * 512 + h * 64;
  const float* a = al + h * 64;
  const float* b = ar + h * 64;
  float sl = 0.f, sr = 0.f;
#pragma unroll 4
  for (int d = 0; d < 64; ++d) { float v = f[d]; sl += v * a[d]; sr += v * b[d]; }
  el[t] = sl; er[t] = sr;
}

__global__ void gat_edge_max(const int* __restrict__ src, const int* __restrict__ dst,
                             const float* __restrict__ el, const float* __restrict__ er,
                             float* __restrict__ e, float* __restrict__ m) {
  int t = blockIdx.x * blockDim.x + threadIdx.x;   // NE*8
  if (t >= NE * 8) return;
  int eid = t >> 3, h = t & 7;
  int s = src[eid], d = dst[eid];
  float x = el[s * 8 + h] + er[d * 8 + h];
  x = x > 0.f ? x : NEG_SLOPE * x;                 // leaky relu
  e[t] = x;
  atomicMaxF32(&m[d * 8 + h], x);
}

__global__ void gat_edge_expsum(const int* __restrict__ dst, float* __restrict__ e,
                                const float* __restrict__ m, float* __restrict__ s) {
  int t = blockIdx.x * blockDim.x + threadIdx.x;   // NE*8
  if (t >= NE * 8) return;
  int eid = t >> 3, h = t & 7;
  int d = dst[eid];
  float ex = __expf(e[t] - m[d * 8 + h]);
  e[t] = ex;
  atomicAdd(&s[d * 8 + h], ex);
}

__global__ void gat_edge_aggr(const int* __restrict__ src, const int* __restrict__ dst,
                              const float* __restrict__ e, const float* __restrict__ s,
                              const float* __restrict__ ft, float* __restrict__ rst) {
  int t = blockIdx.x * blockDim.x + threadIdx.x;   // NE*128 (8 heads x 16 float4 chunks)
  if (t >= NE * 128) return;
  int eid = t >> 7;
  int r   = t & 127;
  int h   = r >> 4;
  int d4  = (r & 15) << 2;
  int sn = src[eid], dn = dst[eid];
  float alpha = e[eid * 8 + h] / s[dn * 8 + h];
  const float4 fv = *(const float4*)(ft + (size_t)sn * 512 + h * 64 + d4);
  float* rp = rst + (size_t)dn * 512 + h * 64 + d4;
  atomicAdd(rp + 0, alpha * fv.x);
  atomicAdd(rp + 1, alpha * fv.y);
  atomicAdd(rp + 2, alpha * fv.z);
  atomicAdd(rp + 3, alpha * fv.w);
}

__global__ void gat_epilogue(const float* __restrict__ rst, const float* __restrict__ bias,
                             float* __restrict__ h, unsigned short* __restrict__ hbf,
                             int residual) {
  int t = blockIdx.x * blockDim.x + threadIdx.x;   // NN*512
  if (t >= NN * 512) return;
  int c = t & 511;
  float v = rst[t] + bias[c];
  if (residual) v += h[t];
  v = v > 0.f ? v : (__expf(v) - 1.f);             // ELU
  h[t] = v;
  hbf[t] = f2bf(v);
}

// ---------------- layer 3 (H=6, D=1) + pooling + readout ----------------

__global__ __launch_bounds__(96)
void gat3_proj(const float* __restrict__ h, const float* __restrict__ W3,
               const float* __restrict__ resW3, float* __restrict__ ft3,
               float* __restrict__ res3) {
  __shared__ float sh[512];
  __shared__ float part[96];
  int n = blockIdx.x;
  for (int i = threadIdx.x; i < 512; i += 96) sh[i] = h[(size_t)n * 512 + i];
  __syncthreads();
  int j = threadIdx.x % 12;     // 0..5 -> W3, 6..11 -> resW3
  int p = threadIdx.x / 12;     // 0..7 partial id
  float acc = 0.f;
  if (p < 8) {
    int jj = (j < 6) ? j : (j - 6);
    const float* w = (j < 6) ? W3 : resW3;
    for (int k = p; k < 512; k += 8) acc += sh[k] * w[k * 6 + jj];
  }
  part[threadIdx.x] = acc;
  __syncthreads();
  if (threadIdx.x < 12) {
    float sum = 0.f;
    for (int q = 0; q < 8; ++q) sum += part[q * 12 + threadIdx.x];
    if (threadIdx.x < 6) ft3[n * 6 + threadIdx.x] = sum;
    else                 res3[n * 6 + threadIdx.x - 6] = sum;
  }
}

__global__ void gat3_scores(const float* __restrict__ ft3, const float* __restrict__ al3,
                            const float* __restrict__ ar3, float* __restrict__ el3,
                            float* __restrict__ er3) {
  int t = blockIdx.x * blockDim.x + threadIdx.x;   // NN*6
  if (t >= NN * 6) return;
  int h = t % 6;
  float v = ft3[t];
  el3[t] = v * al3[h];
  er3[t] = v * ar3[h];
}

__global__ void gat3_edge_max(const int* __restrict__ src, const int* __restrict__ dst,
                              const float* __restrict__ el3, const float* __restrict__ er3,
                              float* __restrict__ e3, float* __restrict__ m3) {
  int t = blockIdx.x * blockDim.x + threadIdx.x;   // NE*6
  if (t >= NE * 6) return;
  int eid = t / 6, h = t % 6;
  float x = el3[src[eid] * 6 + h] + er3[dst[eid] * 6 + h];
  x = x > 0.f ? x : NEG_SLOPE * x;
  e3[t] = x;
  atomicMaxF32(&m3[dst[eid] * 6 + h], x);
}

__global__ void gat3_edge_expsum(const int* __restrict__ dst, float* __restrict__ e3,
                                 const float* __restrict__ m3, float* __restrict__ s3) {
  int t = blockIdx.x * blockDim.x + threadIdx.x;   // NE*6
  if (t >= NE * 6) return;
  int eid = t / 6, h = t % 6;
  float ex = __expf(e3[t] - m3[dst[eid] * 6 + h]);
  e3[t] = ex;
  atomicAdd(&s3[dst[eid] * 6 + h], ex);
}

__global__ void gat3_edge_aggr(const int* __restrict__ src, const int* __restrict__ dst,
                               const float* __restrict__ e3, const float* __restrict__ s3,
                               const float* __restrict__ ft3, float* __restrict__ rst3) {
  int t = blockIdx.x * blockDim.x + threadIdx.x;   // NE*6
  if (t >= NE * 6) return;
  int eid = t / 6, h = t % 6;
  int dn = dst[eid];
  float alpha = e3[t] / s3[dn * 6 + h];
  atomicAdd(&rst3[dn * 6 + h], alpha * ft3[src[eid] * 6 + h]);
}

__global__ void gat3_pool(const float* __restrict__ rst3, const float* __restrict__ res3,
                          const float* __restrict__ b3, const int* __restrict__ gid,
                          float* __restrict__ pooled) {
  int t = blockIdx.x * blockDim.x + threadIdx.x;   // NN*6
  if (t >= NN * 6) return;
  int n = t / 6, h = t % 6;
  float v = rst3[t] + res3[t] + b3[h];
  atomicAdd(&pooled[gid[n] * 6 + h], v);
}

__global__ void gat_readout(const float* __restrict__ pooled, const float* __restrict__ linW,
                            const float* __restrict__ linb, float* __restrict__ out) {
  int g = threadIdx.x;
  if (g >= NG) return;
  float acc = linb[0];
#pragma unroll
  for (int h = 0; h < 6; ++h) acc += pooled[g * 6 + h] * linW[h];
  out[g] = acc;
}

// ---------------- host orchestration ----------------

extern "C" void kernel_launch(void* const* d_in, const int* in_sizes, int n_in,
                              void* d_out, int out_size, void* d_ws, size_t ws_size,
                              hipStream_t stream) {
  const float* feat  = (const float*)d_in[0];
  const int*   src   = (const int*)d_in[1];
  const int*   dst   = (const int*)d_in[2];
  const int*   gid   = (const int*)d_in[3];
  const float* W0    = (const float*)d_in[4];
  const float* al0   = (const float*)d_in[5];
  const float* ar0   = (const float*)d_in[6];
  const float* b0    = (const float*)d_in[7];
  const float* W1    = (const float*)d_in[8];
  const float* al1   = (const float*)d_in[9];
  const float* ar1   = (const float*)d_in[10];
  const float* b1    = (const float*)d_in[11];
  const float* W2    = (const float*)d_in[12];
  const float* al2   = (const float*)d_in[13];
  const float* ar2   = (const float*)d_in[14];
  const float* b2    = (const float*)d_in[15];
  const float* W3    = (const float*)d_in[16];
  const float* al3   = (const float*)d_in[17];
  const float* ar3   = (const float*)d_in[18];
  const float* b3    = (const float*)d_in[19];
  const float* resW3 = (const float*)d_in[20];
  const float* linW  = (const float*)d_in[21];
  const float* linb  = (const float*)d_in[22];
  float* out = (float*)d_out;

  size_t off = 0;
  auto wsAlloc = [&](size_t bytes) -> void* {
    void* p = (char*)d_ws + off;
    off += (bytes + 255) & ~(size_t)255;
    return p;
  };
  float* ft   = (float*)wsAlloc((size_t)NN * 512 * 4);
  float* hbuf = (float*)wsAlloc((size_t)NN * 512 * 4);
  float* rst  = (float*)wsAlloc((size_t)NN * 512 * 4);
  unsigned short* hbf    = (unsigned short*)wsAlloc((size_t)NN * 512 * 2);
  unsigned short* featbf = (unsigned short*)wsAlloc((size_t)NN * 128 * 2);
  unsigned short* wt0 = (unsigned short*)wsAlloc((size_t)512 * 128 * 2);
  unsigned short* wt1 = (unsigned short*)wsAlloc((size_t)512 * 512 * 2);
  unsigned short* wt2 = (unsigned short*)wsAlloc((size_t)512 * 512 * 2);
  float* el   = (float*)wsAlloc((size_t)NN * 8 * 4);
  float* er   = (float*)wsAlloc((size_t)NN * 8 * 4);
  float* mbuf = (float*)wsAlloc((size_t)NN * 8 * 4);
  float* sbuf = (float*)wsAlloc((size_t)NN * 8 * 4);
  float* ebuf = (float*)wsAlloc((size_t)NE * 8 * 4);
  float* ft3  = (float*)wsAlloc((size_t)NN * 6 * 4);
  float* res3 = (float*)wsAlloc((size_t)NN * 6 * 4);
  float* el3  = (float*)wsAlloc((size_t)NN * 6 * 4);
  float* er3  = (float*)wsAlloc((size_t)NN * 6 * 4);
  float* m3   = (float*)wsAlloc((size_t)NN * 6 * 4);
  float* s3   = (float*)wsAlloc((size_t)NN * 6 * 4);
  float* rst3 = (float*)wsAlloc((size_t)NN * 6 * 4);
  float* e3   = (float*)wsAlloc((size_t)NE * 6 * 4);
  float* pooled = (float*)wsAlloc((size_t)NG * 6 * 4);
  (void)ws_size; (void)n_in; (void)in_sizes; (void)out_size;

  const int T = 256;
  auto cdiv = [](int a, int b) { return (a + b - 1) / b; };
  const float NEGINF = -__builtin_huge_valf();

  // one-time converts
  conv_bf16<<<cdiv(NN * 128, T), T, 0, stream>>>(feat, featbf, NN * 128);
  convT_bf16<<<cdiv(128 * 512, T), T, 0, stream>>>(W0, wt0, 128, 512);
  convT_bf16<<<cdiv(512 * 512, T), T, 0, stream>>>(W1, wt1, 512, 512);
  convT_bf16<<<cdiv(512 * 512, T), T, 0, stream>>>(W2, wt2, 512, 512);

  const unsigned short* gemmA[3] = { featbf, hbf, hbf };
  const unsigned short* gemmB[3] = { wt0, wt1, wt2 };
  const int   gemmK[3]  = { 128, 512, 512 };
  const float* als[3]   = { al0, al1, al2 };
  const float* ars[3]   = { ar0, ar1, ar2 };
  const float* bs[3]    = { b0, b1, b2 };

  for (int layer = 0; layer < 3; ++layer) {
    gat_gemm_bf16<<<dim3(NN / 128, 512 / 64), T, 0, stream>>>(gemmA[layer], gemmB[layer], ft, gemmK[layer]);
    gat_scores<<<cdiv(NN * 8, T), T, 0, stream>>>(ft, als[layer], ars[layer], el, er);
    fill_f32<<<cdiv(NN * 8, T), T, 0, stream>>>(mbuf, NEGINF, NN * 8);
    fill_f32<<<cdiv(NN * 8, T), T, 0, stream>>>(sbuf, 0.f, NN * 8);
    fill_f32<<<cdiv(NN * 512, T), T, 0, stream>>>(rst, 0.f, NN * 512);
    gat_edge_max<<<cdiv(NE * 8, T), T, 0, stream>>>(src, dst, el, er, ebuf, mbuf);
    gat_edge_expsum<<<cdiv(NE * 8, T), T, 0, stream>>>(dst, ebuf, mbuf, sbuf);
    gat_edge_aggr<<<cdiv(NE * 128, T), T, 0, stream>>>(src, dst, ebuf, sbuf, ft, rst);
    gat_epilogue<<<cdiv(NN * 512, T), T, 0, stream>>>(rst, bs[layer], hbuf, hbf, layer > 0 ? 1 : 0);
  }

  // layer 3
  gat3_proj<<<NN, 96, 0, stream>>>(hbuf, W3, resW3, ft3, res3);
  gat3_scores<<<cdiv(NN * 6, T), T, 0, stream>>>(ft3, al3, ar3, el3, er3);
  fill_f32<<<cdiv(NN * 6, T), T, 0, stream>>>(m3, NEGINF, NN * 6);
  fill_f32<<<cdiv(NN * 6, T), T, 0, stream>>>(s3, 0.f, NN * 6);
  fill_f32<<<cdiv(NN * 6, T), T, 0, stream>>>(rst3, 0.f, NN * 6);
  fill_f32<<<cdiv(NG * 6, T), T, 0, stream>>>(pooled, 0.f, NG * 6);
  gat3_edge_max<<<cdiv(NE * 6, T), T, 0, stream>>>(src, dst, el3, er3, e3, m3);
  gat3_edge_expsum<<<cdiv(NE * 6, T), T, 0, stream>>>(dst, e3, m3, s3);
  gat3_edge_aggr<<<cdiv(NE * 6, T), T, 0, stream>>>(src, dst, e3, s3, ft3, rst3);
  gat3_pool<<<cdiv(NN * 6, T), T, 0, stream>>>(rst3, res3, b3, gid, pooled);
  gat_readout<<<1, 64, 0, stream>>>(pooled, linW, linb, out);
}
